// MGTABModel_63273458205158
// MI455X (gfx1250) — compile-verified, matrix operand order checked by reading, not run on hardware
//
#include <hip/hip_runtime.h>
#include <hip/hip_bf16.h>

typedef __bf16 bf16_t;
typedef __attribute__((ext_vector_type(16))) __bf16 v16bf;
typedef __attribute__((ext_vector_type(8)))  float  v8f;
typedef unsigned int u32x4 __attribute__((ext_vector_type(4)));
typedef int          i32x8 __attribute__((ext_vector_type(8)));
typedef int          i32x4 __attribute__((ext_vector_type(4)));

#define D_FEAT 128
#define NREL   7
#define NMAT   8                       // 7 relations + root
#define MAT_ELEMS (4 * 8 * 32 * 16)    // one matrix in fragment order: 16384 bf16 = 32KB
#define PACKED_ELEMS ((size_t)NMAT * MAT_ELEMS)

// ---------------------------------------------------------------------------
// zero fill
// ---------------------------------------------------------------------------
__global__ void zero_f32(float* __restrict__ p, size_t n) {
    size_t i = (size_t)blockIdx.x * blockDim.x + threadIdx.x;
    if (i < n) p[i] = 0.0f;
}

// ---------------------------------------------------------------------------
// edge scatter: sums[(dst*NREL + etype)*128 + t] += feat[src*128 + t]
// ---------------------------------------------------------------------------
__global__ void scatter_edges(const float* __restrict__ feat,
                              const int* __restrict__ src,
                              const int* __restrict__ dst,
                              const int* __restrict__ etype,
                              float* __restrict__ sums, int n_edges) {
    size_t i = (size_t)blockIdx.x * blockDim.x + threadIdx.x;
    size_t total = (size_t)n_edges * D_FEAT;
    if (i >= total) return;
    int e = (int)(i >> 7);
    int t = (int)(i & 127);
    int s = src[e], d = dst[e], r = etype[e];
    float v = feat[(size_t)s * D_FEAT + t];
    atomicAdd(sums + ((size_t)d * NREL + r) * D_FEAT + t, v);
}

__global__ void scatter_counts(const int* __restrict__ dst,
                               const int* __restrict__ etype,
                               float* __restrict__ cnt, int n_edges) {
    int e = blockIdx.x * blockDim.x + threadIdx.x;
    if (e >= n_edges) return;
    atomicAdd(cnt + (size_t)dst[e] * NREL + etype[e], 1.0f);
}

// ---------------------------------------------------------------------------
// pack fp32 weights into bf16 WMMA B-fragment order.
// B fragment (16-bit, 32x16): lane l holds column N = l&15;
//   K values: k0 = kt*32 + ((l&16)?8:0); half h -> K = k0 + (h&7) + (h>=8?16:0)
// ---------------------------------------------------------------------------
__global__ void pack_weights(const float* __restrict__ Wrel,   // [7][128][128]
                             const float* __restrict__ Wroot,  // [128][128]
                             bf16_t* __restrict__ packed) {
    int idx = blockIdx.x * blockDim.x + threadIdx.x;  // NMAT*4*8*32 = 8192
    if (idx >= NMAT * 4 * 8 * 32) return;
    int lane = idx & 31;
    int nt   = (idx >> 5) & 7;
    int kt   = (idx >> 8) & 3;
    int mat  = idx >> 10;
    const float* W = (mat < NREL) ? (Wrel + (size_t)mat * D_FEAT * D_FEAT) : Wroot;
    int n  = nt * 16 + (lane & 15);
    int k0 = kt * 32 + ((lane & 16) ? 8 : 0);
    bf16_t* dstp = packed + (size_t)idx * 16;
#pragma unroll
    for (int h = 0; h < 16; ++h) {
        int k = k0 + (h & 7) + ((h >= 8) ? 16 : 0);
        dstp[h] = (bf16_t)W[(size_t)k * D_FEAT + n];
    }
}

// ---------------------------------------------------------------------------
// TDM: async DMA one 32KB fragment-ordered matrix slab (4096 x 8B) into LDS.
// D# per CDNA5 ISA ch.8; tracked by TENSORcnt; TDM ignores EXEC -> one wave issues.
// ---------------------------------------------------------------------------
#if __has_builtin(__builtin_amdgcn_tensor_load_to_lds)
#define HAVE_TDM 1
__device__ inline void tdm_load_slab(const bf16_t* gsrc, bf16_t* ldst) {
    unsigned long long ga = (unsigned long long)(uintptr_t)gsrc;
    unsigned int glo = (unsigned int)ga;
    unsigned int ghi = (unsigned int)(ga >> 32);
    unsigned int lds_off = (unsigned int)(uintptr_t)ldst;  // flat shared addr[31:0] = LDS byte offset
    u32x4 g0 = { 1u,                                    // count=1 (valid user D#)
                 lds_off,                               // lds_addr (bytes)
                 glo,                                   // global_addr[31:0]
                 (ghi & 0x1FFFFFFu) | (2u << 30) };     // global_addr[56:32] | type=2
    i32x8 g1 = { (int)(3u << 16),                       // data_size = 8 bytes
                 (int)(4096u << 16),                    // tensor_dim0[15:0] << 16
                 (int)(1u << 16),                       // tensor_dim0 hi | tensor_dim1=1
                 (int)(4096u << 16),                    // tensor_dim1 hi | tile_dim0=4096
                 0,                                     // tile_dim1=0, tile_dim2=0 (unused)
                 4096,                                  // tensor_dim0_stride[31:0]
                 0, 0 };
    i32x4 g2 = { 1, 1, 4096, 0 };                       // tensor_dim2=1, tensor_dim3=1, dim2_stride
    i32x4 g3 = { 4096, (int)(1u << 16), 0, 0 };         // dim3_stride, tensor_dim4=1, tile_dim4=0
    i32x8 gx = { 0, 0, 0, 0, 0, 0, 0, 0 };              // aux group (6-arg form), unused
    __builtin_amdgcn_tensor_load_to_lds(g0, g1, g2, g3, gx, 0);
}
#else
#define HAVE_TDM 0
#endif

// ---------------------------------------------------------------------------
// A-path software pipeline: raw row loads (16 x b128, clause-able) kept in
// registers one matrix ahead; scale fused into the f32->bf16 convert.
// ---------------------------------------------------------------------------
struct ARaw { float4 c[16]; };

__device__ inline void load_araw(const float* __restrict__ rowbase, int lane, ARaw& ar) {
    const int k0 = (lane & 16) ? 8 : 0;
#pragma unroll
    for (int kt = 0; kt < 4; ++kt) {
        const float4* p = (const float4*)(rowbase + kt * 32 + k0);
        const float4* q = (const float4*)(rowbase + kt * 32 + k0 + 16);
        ar.c[kt * 4 + 0] = p[0];
        ar.c[kt * 4 + 1] = p[1];
        ar.c[kt * 4 + 2] = q[0];
        ar.c[kt * 4 + 3] = q[1];
    }
}

__device__ inline void convert_afrags(const ARaw& ar, float scale, v16bf* afrag) {
#pragma unroll
    for (int kt = 0; kt < 4; ++kt) {
        v16bf a;
#pragma unroll
        for (int i = 0; i < 4; ++i) {
            float4 v = ar.c[kt * 4 + i];
            a[i * 4 + 0] = (bf16_t)(v.x * scale);
            a[i * 4 + 1] = (bf16_t)(v.y * scale);
            a[i * 4 + 2] = (bf16_t)(v.z * scale);
            a[i * 4 + 3] = (bf16_t)(v.w * scale);
        }
        afrag[kt] = a;
    }
}

__device__ inline const float* mat_rowbase(int mat, int nodec,
                                           const float* __restrict__ sums,
                                           const float* __restrict__ feat,
                                           const float* __restrict__ cnt,
                                           float inb, float& scale) {
    if (mat < NREL) {
        size_t seg = (size_t)nodec * NREL + mat;
        scale = inb / fmaxf(cnt[seg], 1.0f);        // segment mean + OOB zeroing
        return sums + seg * D_FEAT;
    }
    scale = inb;                                    // root term
    return feat + (size_t)nodec * D_FEAT;
}

// ---------------------------------------------------------------------------
// RGCN layer GEMM: out[node] = relu( sum_r mean[node,r] @ W_r + feat @ W_root + b )
// one wave per 16-node tile; 8 waves/block share TDM-staged weights in LDS,
// double-buffered; A rows and B fragments software-pipelined one step ahead.
// ---------------------------------------------------------------------------
__global__ __launch_bounds__(256) void rgcn_gemm_wmma(
    const float* __restrict__ feat,     // [N][128]
    const float* __restrict__ sums,     // [N][NREL][128]
    const float* __restrict__ cnt,      // [N][NREL]
    const bf16_t* __restrict__ packedW, // fragment-ordered, NMAT matrices
    const float* __restrict__ bias,     // [128]
    float* __restrict__ out,            // [N][128]
    int n_nodes) {
    __shared__ __align__(32) bf16_t lds_w[2][MAT_ELEMS];   // 2 x 32KB

    const int warp = threadIdx.x >> 5;
    const int lane = threadIdx.x & 31;
    const int tile = blockIdx.x * 8 + warp;
    const int m    = lane & 15;
    const int hi   = (lane & 16) ? 8 : 0;
    const int node = tile * 16 + m;
    const float inb  = (node < n_nodes) ? 1.0f : 0.0f;
    const int  nodec = (node < n_nodes) ? node : (n_nodes - 1);

    v8f acc[8];
#pragma unroll
    for (int nt = 0; nt < 8; ++nt) {
        float bv = bias[nt * 16 + m];
#pragma unroll
        for (int j = 0; j < 8; ++j) acc[nt][j] = bv;
    }

#if HAVE_TDM
    if (warp == 0) tdm_load_slab(packedW, &lds_w[0][0]);       // prefetch B slab for mat 0
#endif

    // prefetch A row for mat 0
    ARaw araw;
    float scale_cur, scale_nxt;
    load_araw(mat_rowbase(0, nodec, sums, feat, cnt, inb, scale_cur), lane, araw);

#pragma unroll 1
    for (int mat = 0; mat < NMAT; ++mat) {
#if HAVE_TDM
        if (warp == 0) {
            if (mat < NMAT - 1) {                               // prefetch next B slab
                tdm_load_slab(packedW + (size_t)(mat + 1) * MAT_ELEMS,
                              &lds_w[(mat + 1) & 1][0]);
                __builtin_amdgcn_s_wait_tensorcnt(1);           // current slab done
            } else {
                __builtin_amdgcn_s_wait_tensorcnt(0);
            }
        }
        __syncthreads();                                        // slab visible to all waves
#else
        {   // fallback: cooperative global->LDS copy of current slab
            const uint4* gs = (const uint4*)(packedW + (size_t)mat * MAT_ELEMS);
            uint4* ls = (uint4*)&lds_w[mat & 1][0];
#pragma unroll
            for (int c = 0; c < (MAT_ELEMS * 2 / 16) / 256; ++c)
                ls[c * 256 + threadIdx.x] = gs[c * 256 + threadIdx.x];
            __syncthreads();
        }
#endif
        // consume prefetched A row for this matrix
        v16bf afrag[4];
        convert_afrags(araw, scale_cur, afrag);

        // issue next matrix's A-row loads; latency hides under the 32 WMMAs
        if (mat < NMAT - 1) {
            load_araw(mat_rowbase(mat + 1, nodec, sums, feat, cnt, inb, scale_nxt),
                      lane, araw);
        }

        // ---- 32 WMMAs, B fragments rotated one ahead through LDS loads ----
        const v16bf* bw = (const v16bf*)&lds_w[mat & 1][0];
        v16bf bcur = bw[lane];                                  // fragment 0
#pragma unroll
        for (int f = 0; f < 32; ++f) {
            v16bf bnext;
            if (f < 31) bnext = bw[(f + 1) * 32 + lane];        // load-ahead
            const int kt = f >> 3;
            const int nt = f & 7;
            acc[nt] = __builtin_amdgcn_wmma_f32_16x16x32_bf16(
                false, afrag[kt], false, bcur, (short)0, acc[nt], false, false);
            bcur = bnext;
        }
        scale_cur = scale_nxt;
        __syncthreads();                                        // done reading before overwrite
    }

    // D layout: VGPR j -> row M = j + hi; column = nt*16 + (lane&15)
#pragma unroll
    for (int j = 0; j < 8; ++j) {
        int nd = tile * 16 + j + hi;
        if (nd < n_nodes) {
            float* orow = out + (size_t)nd * D_FEAT + m;
#pragma unroll
            for (int nt = 0; nt < 8; ++nt)
                orow[nt * 16] = fmaxf(acc[nt][j], 0.0f);
        }
    }
}

// ---------------------------------------------------------------------------
// multi-task head: bot_logits (N,) then stance_logits (N,3)
// ---------------------------------------------------------------------------
__global__ void head_kernel(const float* __restrict__ h,
                            const float* __restrict__ bot_w, const float* __restrict__ bot_b,
                            const float* __restrict__ st_w,  const float* __restrict__ st_b,
                            float* __restrict__ out, int n) {
    int i = blockIdx.x * blockDim.x + threadIdx.x;
    if (i >= n) return;
    const float* row = h + (size_t)i * D_FEAT;
    float a0 = 0.f, a1 = 0.f, a2 = 0.f, a3 = 0.f;
#pragma unroll 4
    for (int k = 0; k < D_FEAT; ++k) {
        float v = fmaxf(row[k], 0.0f);
        a0 += v * bot_w[k];
        a1 += v * st_w[k * 3 + 0];
        a2 += v * st_w[k * 3 + 1];
        a3 += v * st_w[k * 3 + 2];
    }
    out[i] = a0 + bot_b[0];
    out[(size_t)n + (size_t)i * 3 + 0] = a1 + st_b[0];
    out[(size_t)n + (size_t)i * 3 + 1] = a2 + st_b[1];
    out[(size_t)n + (size_t)i * 3 + 2] = a3 + st_b[2];
}

// ---------------------------------------------------------------------------
extern "C" void kernel_launch(void* const* d_in, const int* in_sizes, int n_in,
                              void* d_out, int out_size, void* d_ws, size_t ws_size,
                              hipStream_t stream) {
    const float* x      = (const float*)d_in[0];
    const int*   ei     = (const int*)d_in[1];
    const int*   etype  = (const int*)d_in[2];
    const float* Wrel1  = (const float*)d_in[3];
    const float* Wroot1 = (const float*)d_in[4];
    const float* b1     = (const float*)d_in[5];
    const float* Wrel2  = (const float*)d_in[6];
    const float* Wroot2 = (const float*)d_in[7];
    const float* b2     = (const float*)d_in[8];
    const float* bot_w  = (const float*)d_in[9];
    const float* bot_b  = (const float*)d_in[10];
    const float* st_w   = (const float*)d_in[11];
    const float* st_b   = (const float*)d_in[12];

    const int n_nodes = in_sizes[0] / D_FEAT;
    const int n_edges = in_sizes[2];
    const int* src = ei;
    const int* dst = ei + n_edges;

    size_t off = 0;
    auto take = [&](size_t bytes) {
        void* p = (char*)d_ws + off;
        off += (bytes + 255) & ~(size_t)255;
        return p;
    };
    size_t sums_elems = (size_t)n_nodes * NREL * D_FEAT;
    size_t cnt_elems  = (size_t)n_nodes * NREL;
    size_t h_elems    = (size_t)n_nodes * D_FEAT;
    float*  sums = (float*)take(sums_elems * sizeof(float));
    float*  cnt  = (float*)take(cnt_elems * sizeof(float));
    float*  h1   = (float*)take(h_elems * sizeof(float));
    float*  h2   = (float*)take(h_elems * sizeof(float));
    bf16_t* pw   = (bf16_t*)take(PACKED_ELEMS * sizeof(bf16_t));

    const int ntiles = (n_nodes + 15) / 16;
    const dim3 gemm_grid((ntiles + 7) / 8);
    const size_t scat_total = (size_t)n_edges * D_FEAT;

    for (int layer = 0; layer < 2; ++layer) {
        const float* feat  = (layer == 0) ? x : h1;
        const float* Wrel  = (layer == 0) ? Wrel1 : Wrel2;
        const float* Wroot = (layer == 0) ? Wroot1 : Wroot2;
        const float* bias  = (layer == 0) ? b1 : b2;
        float* hout        = (layer == 0) ? h1 : h2;

        zero_f32<<<(unsigned)((sums_elems + 255) / 256), 256, 0, stream>>>(sums, sums_elems);
        zero_f32<<<(unsigned)((cnt_elems + 255) / 256), 256, 0, stream>>>(cnt, cnt_elems);
        scatter_edges<<<(unsigned)((scat_total + 255) / 256), 256, 0, stream>>>(
            feat, src, dst, etype, sums, n_edges);
        scatter_counts<<<(n_edges + 255) / 256, 256, 0, stream>>>(dst, etype, cnt, n_edges);
        pack_weights<<<(NMAT * 4 * 8 * 32 + 255) / 256, 256, 0, stream>>>(Wrel, Wroot, pw);
        rgcn_gemm_wmma<<<gemm_grid, 256, 0, stream>>>(feat, sums, cnt, pw, bias, hout, n_nodes);
    }

    head_kernel<<<(n_nodes + 255) / 256, 256, 0, stream>>>(h2, bot_w, bot_b, st_w, st_b,
                                                           (float*)d_out, n_nodes);
}
